// NonLocalBlock_35347580846181
// MI455X (gfx1250) — matrix operand order repaired and checked by reading, not compiled, not run on hardware
//
#include <hip/hip_runtime.h>

typedef __attribute__((ext_vector_type(16))) __bf16 v16bf;
typedef __attribute__((ext_vector_type(8)))  float  v8f;
typedef __attribute__((ext_vector_type(4)))  unsigned int v4u;

#define NB  4
#define NC  64
#define NCI 32
#define NN  4096  // H*W

#define WMMA_BF16(A, B, C)                                                   \
  __builtin_amdgcn_wmma_f32_16x16x32_bf16(false, (A), false, (B), (short)0,  \
                                          (C), false, false)

// Load two 16-byte runs into one 8-VGPR bf16 operand (elements 0-7 / 8-15).
__device__ __forceinline__ v16bf ld2x16(const __bf16* p0, const __bf16* p1) {
  union { v4u q[2]; v16bf v; } u;
  u.q[0] = *(const v4u*)p0;
  u.q[1] = *(const v4u*)p1;
  return u.v;
}

// -------------------------------------------------------------------------
// Prep 1: split-transpose x -> xT_hi/xT_lo [b][n][c] (bf16), hi+lo ~ fp32.
// -------------------------------------------------------------------------
__global__ __launch_bounds__(256) void k_prep_x(const float* __restrict__ x,
                                                __bf16* __restrict__ xh,
                                                __bf16* __restrict__ xl) {
  int t = blockIdx.x * 256 + threadIdx.x;  // (b, n)
  int b = t >> 12, n = t & (NN - 1);
  const float* xb = x + ((size_t)b * NC) * NN + n;
  __bf16* ph = xh + ((size_t)b * NN + n) * NC;
  __bf16* pl = xl + ((size_t)b * NN + n) * NC;
#pragma unroll
  for (int c = 0; c < NC; ++c) {
    float v  = xb[(size_t)c * NN];
    __bf16 h = (__bf16)v;
    ph[c] = h;
    pl[c] = (__bf16)(v - (float)h);
  }
}

// -------------------------------------------------------------------------
// Prep 2: g[b,n,o] = x[b,:,n]·Wg[o,:] + bg[o], stored split-transposed
//         gT_hi/gT_lo [b][ci][n] (bf16).
// -------------------------------------------------------------------------
__global__ __launch_bounds__(256) void k_gproj(const float* __restrict__ x,
                                               const float* __restrict__ Wg,
                                               const float* __restrict__ bg,
                                               __bf16* __restrict__ gh,
                                               __bf16* __restrict__ gl) {
  int t = blockIdx.x * 256 + threadIdx.x;  // (b, n)
  int b = t >> 12, n = t & (NN - 1);
  const float* xb = x + ((size_t)b * NC) * NN + n;
  float xr[NC];
#pragma unroll
  for (int c = 0; c < NC; ++c) xr[c] = xb[(size_t)c * NN];
#pragma unroll 4
  for (int o = 0; o < NCI; ++o) {
    float acc = bg[o];
#pragma unroll
    for (int c = 0; c < NC; ++c) acc += xr[c] * Wg[o * NC + c];
    __bf16 h = (__bf16)acc;
    gh[((size_t)b * NCI + o) * NN + n] = h;
    gl[((size_t)b * NCI + o) * NN + n] = (__bf16)(acc - (float)h);
  }
}

// -------------------------------------------------------------------------
// Kernel 2: streaming attention, bf16x3 split precision on
// v_wmma_f32_16x16x32_bf16.
//   Per wave: 16 queries. Loop over 128 chunks of 32 keys:
//     S^T (2 tiles, 6 WMMA each: Ahi*Bhi + Ahi*Blo + Alo*Bhi over K=64)
//     P = exp(S^T)  -> packs straight into the 16-bit A layout (no shuffles:
//                      A-layout half-split {0-7,16-23}/{8-15,24-31} == D rows v+8h)
//     Yacc += P*G   (2 ci-tiles x 3 split terms, K=32)
//   Final: y[b,ci,q0+m] = Yacc[m,ci] / rowsum.
// -------------------------------------------------------------------------
__global__ __launch_bounds__(128) void k_attn(const __bf16* __restrict__ xh,
                                              const __bf16* __restrict__ xl,
                                              const __bf16* __restrict__ gh,
                                              const __bf16* __restrict__ gl,
                                              float* __restrict__ y) {
  const int lane = threadIdx.x & 31;
  const int wid  = threadIdx.x >> 5;
  const int l16  = lane & 15;
  const int hh   = lane >> 4;  // half-wave id
  const int b    = blockIdx.y;
  const int q0   = blockIdx.x * 64 + wid * 16;

  const __bf16* xhb = xh + ((size_t)b * NN) * NC;
  const __bf16* xlb = xl + ((size_t)b * NN) * NC;
  const __bf16* ghb = gh + ((size_t)b * NCI) * NN;
  const __bf16* glb = gl + ((size_t)b * NCI) * NN;

  // Query B-operands (register resident): B[c,m], lane m=l16 holds
  // c = ch*32 + 8*hh + [0..7], ch*32 + 16 + 8*hh + [0..7].
  v16bf bqh[2], bql[2];
#pragma unroll
  for (int ch = 0; ch < 2; ++ch) {
    const __bf16* qh = xhb + (size_t)(q0 + l16) * NC + ch * 32 + 8 * hh;
    const __bf16* ql = xlb + (size_t)(q0 + l16) * NC + ch * 32 + 8 * hh;
    bqh[ch] = ld2x16(qh, qh + 16);
    bql[ch] = ld2x16(ql, ql + 16);
  }

  v8f acc0 = {};  // Y, ci = l16
  v8f acc1 = {};  // Y, ci = l16 + 16
  float lsum = 0.f;

  for (int j = 0; j < NN / 32; ++j) {
    const int n0 = j * 32;
    if (j + 1 < NN / 32) {
      // WGP-scope prefetch (locality 3) actually fills L0/WGP$ per the ISA
      // prefetch-TH table; SYS-scope (locality 0) would stop at GL2, where
      // the 6 MB working set already lives.
      __builtin_prefetch(xhb + (size_t)(n0 + 32 + l16) * NC, 0, 3);
      __builtin_prefetch(ghb + (size_t)(l16 + 16 * hh) * NN + n0 + 32, 0, 3);
    }

    // ---- S^T tiles: keys x queries (3-term bf16 split, K=64) ----
    v8f s[2];
#pragma unroll
    for (int kt = 0; kt < 2; ++kt) {
      const int nk = n0 + kt * 16;
      v8f sv = {};
#pragma unroll
      for (int ch = 0; ch < 2; ++ch) {
        // A[key, c]: lane key=l16; elems 0-7: c = ch*32+8*hh+[0..7],
        //                          elems 8-15: c = ch*32+16+8*hh+[0..7].
        const __bf16* ka = xhb + (size_t)(nk + l16) * NC + ch * 32 + 8 * hh;
        const __bf16* kl = xlb + (size_t)(nk + l16) * NC + ch * 32 + 8 * hh;
        v16bf akh = ld2x16(ka, ka + 16);
        v16bf akl = ld2x16(kl, kl + 16);
        sv = WMMA_BF16(akh, bqh[ch], sv);
        sv = WMMA_BF16(akh, bql[ch], sv);
        sv = WMMA_BF16(akl, bqh[ch], sv);
      }
      s[kt] = sv;
    }

    // ---- exponentiate (no max-sub needed: logits <~ 70, exp fits f32) ----
    float p0[8], p1[8];
#pragma unroll
    for (int v = 0; v < 8; ++v) { p0[v] = __expf(s[0][v]); lsum += p0[v]; }
#pragma unroll
    for (int v = 0; v < 8; ++v) { p1[v] = __expf(s[1][v]); lsum += p1[v]; }

    // ---- pack P into 16-bit A layout: elem e<8 -> key n0+e+8*hh (tile0),
    //      elem e>=8 -> key n0+16+(e-8)+8*hh (tile1). Pure per-lane packing.
    v16bf aph, apl;
#pragma unroll
    for (int e = 0; e < 8; ++e) {
      __bf16 h0 = (__bf16)p0[e];
      aph[e]     = h0;
      apl[e]     = (__bf16)(p0[e] - (float)h0);
      __bf16 h1 = (__bf16)p1[e];
      aph[e + 8] = h1;
      apl[e + 8] = (__bf16)(p1[e] - (float)h1);
    }

    // ---- Yacc += P * G (K = 32 keys, 2 ci-tiles, 3 split terms) ----
    // B[k, ci]: lane ci; elems 0-7: k = n0 + 8*hh + [0..7], 8-15: k = n0 + 16 + 8*hh + [0..7].
    const __bf16* g0h = ghb + (size_t)l16 * NN + n0 + 8 * hh;
    const __bf16* g0l = glb + (size_t)l16 * NN + n0 + 8 * hh;
    const __bf16* g1h = ghb + (size_t)(l16 + 16) * NN + n0 + 8 * hh;
    const __bf16* g1l = glb + (size_t)(l16 + 16) * NN + n0 + 8 * hh;
    v16bf B0h = ld2x16(g0h, g0h + 16), B0l = ld2x16(g0l, g0l + 16);
    v16bf B1h = ld2x16(g1h, g1h + 16), B1l = ld2x16(g1l, g1l + 16);
    acc0 = WMMA_BF16(aph, B0h, acc0);
    acc0 = WMMA_BF16(aph, B0l, acc0);
    acc0 = WMMA_BF16(apl, B0h, acc0);
    acc1 = WMMA_BF16(aph, B1h, acc1);
    acc1 = WMMA_BF16(aph, B1l, acc1);
    acc1 = WMMA_BF16(apl, B1h, acc1);
  }

  // Rowsum for query m=l16: combine the two nc-halves.
  const float Ltot = lsum + __shfl_xor(lsum, 16, 32);

  // acc[v] is Y[m = v + 8*hh, ci]; fetch L[v + 8*hh] via lane broadcast.
  float* yb = y + (size_t)b * NCI * NN;
#pragma unroll
  for (int v = 0; v < 8; ++v) {
    const float Lm  = __shfl(Ltot, v + (hh << 3), 32);
    const float inv = 1.0f / Lm;
    const int row   = q0 + v + (hh << 3);
    yb[(size_t)(l16 + 0)  * NN + row] = acc0[v] * inv;
    yb[(size_t)(l16 + 16) * NN + row] = acc1[v] * inv;
  }
}

// -------------------------------------------------------------------------
// Kernel 3: out[b,o,n] = Wz[o,:]·y[b,:,n] + bz[o] + x[b,o,n]
// -------------------------------------------------------------------------
__global__ __launch_bounds__(256) void k_out(const float* __restrict__ x,
                                             const float* __restrict__ Wz,
                                             const float* __restrict__ bz,
                                             const float* __restrict__ y,
                                             float* __restrict__ out) {
  const int n = blockIdx.x * 256 + threadIdx.x;
  const int o = blockIdx.y;
  const int b = blockIdx.z;
  float acc = bz[o];
#pragma unroll
  for (int i = 0; i < NCI; ++i)
    acc += Wz[o * NCI + i] * y[((size_t)b * NCI + i) * NN + n];
  const size_t idx = ((size_t)b * NC + o) * NN + n;
  out[idx] = acc + x[idx];
}

// -------------------------------------------------------------------------
extern "C" void kernel_launch(void* const* d_in, const int* in_sizes, int n_in,
                              void* d_out, int out_size, void* d_ws, size_t ws_size,
                              hipStream_t stream) {
  const float* x  = (const float*)d_in[0];
  const float* Wg = (const float*)d_in[1];
  const float* bg = (const float*)d_in[2];
  const float* Wz = (const float*)d_in[3];
  const float* bz = (const float*)d_in[4];
  float* out = (float*)d_out;

  // Workspace layout (8 MB total, 16B-aligned sections):
  __bf16* xhw = (__bf16*)d_ws;                   // B*N*C   bf16 (2 MB)
  __bf16* xlw = xhw + (size_t)NB * NN * NC;      // B*N*C   bf16 (2 MB)
  __bf16* ghw = xlw + (size_t)NB * NN * NC;      // B*CI*N  bf16 (1 MB)
  __bf16* glw = ghw + (size_t)NB * NCI * NN;     // B*CI*N  bf16 (1 MB)
  float*  yv  = (float*)(glw + (size_t)NB * NCI * NN);  // B*CI*N f32 (2 MB)

  k_prep_x<<<dim3((NB * NN) / 256), dim3(256), 0, stream>>>(x, xhw, xlw);
  k_gproj <<<dim3((NB * NN) / 256), dim3(256), 0, stream>>>(x, Wg, bg, ghw, glw);
  k_attn  <<<dim3(NN / 64, NB),     dim3(128), 0, stream>>>(xhw, xlw, ghw, glw, yv);
  k_out   <<<dim3(NN / 256, NC, NB), dim3(256), 0, stream>>>(x, Wz, bz, yv, out);
}
